// GlobalAttentionPool_43052752175239
// MI455X (gfx1250) — matrix operand order repaired and checked by reading, not compiled, not run on hardware
//
#include <hip/hip_runtime.h>
#include <stdint.h>

#define NUM_GRAPHS 8192
#define D_IN 256
#define D_HID 128

// LDS row pitches padded by 8 bf16 (16 B) so that 16 lanes reading 16
// consecutive rows hit distinct banks (pitch mod 64 dwords = 4).
#define H_PITCH 264   // D_IN + 8
#define X_PITCH 136   // D_HID + 8

typedef __attribute__((ext_vector_type(16))) __bf16 v16bf;
typedef __attribute__((ext_vector_type(8)))  float  v8f;

static __device__ __forceinline__ uint16_t f2bf_bits(float f) {
  union { float f; uint32_t u; } v; v.f = f;
  uint32_t r = v.u + 0x7FFFu + ((v.u >> 16) & 1u);   // round-to-nearest-even
  return (uint16_t)(r >> 16);
}
static __device__ __forceinline__ __bf16 f2bf(float f) {
  uint16_t b = f2bf_bits(f);
  __bf16 h; __builtin_memcpy(&h, &b, 2);
  return h;
}
static __device__ __forceinline__ float leaky(float v) {
  return v > 0.0f ? v : 0.01f * v;
}
// order-preserving float <-> uint encoding for atomicMax on floats
static __device__ __forceinline__ uint32_t enc_f32(float f) {
  union { float f; uint32_t u; } v; v.f = f;
  return (v.u & 0x80000000u) ? ~v.u : (v.u | 0x80000000u);
}
static __device__ __forceinline__ float dec_f32(uint32_t u) {
  uint32_t b = (u & 0x80000000u) ? (u & 0x7FFFFFFFu) : ~u;
  union { float f; uint32_t u; } v; v.u = b;
  return v.f;
}

// ---------------------------------------------------------------------------
// Swizzle W1/W2 (f32 [K][N] row-major) into bf16 WMMA B-fragment order:
// frag index = ((kstep*8 + ntile)*32 + lane)*16 + e
//   column n = ntile*16 + (lane&15) ;  K = kstep*32 + (e>>3)*16 + (lane>>4)*8 + (e&7)
// so each lane's 16 elements are contiguous (32B) in LDS.
// ---------------------------------------------------------------------------
__global__ void prep_weights_kernel(const float* __restrict__ W1,
                                    const float* __restrict__ W2,
                                    uint16_t* __restrict__ w1f,
                                    uint16_t* __restrict__ w2f) {
  int i = blockIdx.x * blockDim.x + threadIdx.x;
  if (i < 8 * 8 * 32 * 16) {           // W1: 256x128
    int ks = i >> 12, nt = (i >> 9) & 7, lane = (i >> 4) & 31, e = i & 15;
    int K = ks * 32 + (e >> 3) * 16 + (lane >> 4) * 8 + (e & 7);
    int n = nt * 16 + (lane & 15);
    w1f[i] = f2bf_bits(W1[K * D_HID + n]);
  }
  if (i < 4 * 8 * 32 * 16) {           // W2: 128x128
    int ks = i >> 12, nt = (i >> 9) & 7, lane = (i >> 4) & 31, e = i & 15;
    int K = ks * 32 + (e >> 3) * 16 + (lane >> 4) * 8 + (e & 7);
    int n = nt * 16 + (lane & 15);
    w2f[i] = f2bf_bits(W2[K * D_HID + n]);
  }
}

__global__ void seg_init_kernel(uint32_t* __restrict__ segmax,
                                float* __restrict__ segsum) {
  int g = blockIdx.x * blockDim.x + threadIdx.x;
  if (g < NUM_GRAPHS) { segmax[g] = 0u; segsum[g] = 0.0f; }   // 0 < enc(-inf)
}

// ---------------------------------------------------------------------------
// Gate MLP -> per-node logit, using v_wmma_f32_16x16x32_bf16.
// 256 threads (8 waves), 128 rows per block, weights + tiles in LDS.
// ---------------------------------------------------------------------------
__global__ __launch_bounds__(256)
void mlp_logit_kernel(const float* __restrict__ h,
                      const float* __restrict__ b1,
                      const float* __restrict__ b2,
                      const float* __restrict__ W3,
                      const float* __restrict__ b3,
                      const uint16_t* __restrict__ w1f,  // w2f contiguous after
                      float* __restrict__ logit, int N) {
  extern __shared__ char smem[];
  __bf16* sW1 = (__bf16*)smem;           // 32768 elems (64 KB)
  __bf16* sW2 = sW1 + 32768;             // 16384 elems (32 KB)
  __bf16* sH  = sW2 + 16384;             // 128*H_PITCH (66 KB, padded)
  __bf16* sX  = sH + 128 * H_PITCH;      // 128*X_PITCH (34 KB, padded)
  float*  sB1 = (float*)(sX + 128 * X_PITCH);
  float*  sB2 = sB1 + 128;
  float*  sW3 = sB2 + 128;

  const int tid = threadIdx.x;
  const int rowsBase = blockIdx.x * 128;

  // stage pre-swizzled weights: 49152 bf16 = 6144 uint4 (ds_store_b128)
  {
    const uint4* src = (const uint4*)w1f;
    uint4* dst = (uint4*)sW1;
    for (int i = tid; i < 6144; i += 256) dst[i] = src[i];
  }
  if (tid < 128) { sB1[tid] = b1[tid]; sB2[tid] = b2[tid]; sW3[tid] = W3[tid]; }

  // stage h tile: coalesced float4 loads, convert to bf16
  for (int i = tid; i < 128 * 64; i += 256) {
    int r = i >> 6, c4 = (i & 63) << 2;
    int row = rowsBase + r;
    float4 f = make_float4(0.f, 0.f, 0.f, 0.f);
    if (row < N) f = *(const float4*)(h + (size_t)row * D_IN + c4);
    __bf16* d = sH + r * H_PITCH + c4;
    d[0] = f2bf(f.x); d[1] = f2bf(f.y); d[2] = f2bf(f.z); d[3] = f2bf(f.w);
  }
  __syncthreads();

  const int lane = tid & 31;
  const int wave = tid >> 5;
  const int lm   = lane & 15;        // A: row-in-tile / B,C,D: column-in-tile
  const int kg   = lane >> 4;        // lane-half selector
  const int waveRow = wave * 16;

  // ---- layer 1: [16x256]bf16 @ [256x128]bf16 -> [16x128]f32 ----
  v8f acc[8];
#pragma unroll
  for (int nt = 0; nt < 8; ++nt)
#pragma unroll
    for (int r = 0; r < 8; ++r) acc[nt][r] = 0.0f;

#pragma unroll
  for (int ks = 0; ks < 8; ++ks) {
    v16bf a;   // 16-bit A 16x32 layout: lane half kg holds K = kg*8+[0,8) and 16+kg*8+[0,8)
    const __bf16* ap = sH + (waveRow + lm) * H_PITCH + ks * 32 + kg * 8;
#pragma unroll
    for (int e = 0; e < 8; ++e) { a[e] = ap[e]; a[e + 8] = ap[e + 16]; }
#pragma unroll
    for (int nt = 0; nt < 8; ++nt) {
      v16bf bfr;
      const __bf16* bp = sW1 + (((ks * 8 + nt) * 32 + lane) << 4);
#pragma unroll
      for (int e = 0; e < 16; ++e) bfr[e] = bp[e];
      acc[nt] = __builtin_amdgcn_wmma_f32_16x16x32_bf16(
          false, a, false, bfr, (short)0, acc[nt], false, false);
    }
  }
  // bias + leaky, relayout D (row=vgpr, col=lane) -> A (row=lane, K=vgpr) via LDS
#pragma unroll
  for (int nt = 0; nt < 8; ++nt) {
    float bias = sB1[nt * 16 + lm];
#pragma unroll
    for (int r = 0; r < 8; ++r) {
      float v = leaky(acc[nt][r] + bias);
      sX[(waveRow + r + kg * 8) * X_PITCH + nt * 16 + lm] = f2bf(v);
    }
  }
  __syncthreads();

  // ---- layer 2: [16x128]bf16 @ [128x128]bf16 -> [16x128]f32 ----
  v8f acc2[8];
#pragma unroll
  for (int nt = 0; nt < 8; ++nt)
#pragma unroll
    for (int r = 0; r < 8; ++r) acc2[nt][r] = 0.0f;

#pragma unroll
  for (int ks = 0; ks < 4; ++ks) {
    v16bf a;
    const __bf16* ap = sX + (waveRow + lm) * X_PITCH + ks * 32 + kg * 8;
#pragma unroll
    for (int e = 0; e < 8; ++e) { a[e] = ap[e]; a[e + 8] = ap[e + 16]; }
#pragma unroll
    for (int nt = 0; nt < 8; ++nt) {
      v16bf bfr;
      const __bf16* bp = sW2 + (((ks * 8 + nt) * 32 + lane) << 4);
#pragma unroll
      for (int e = 0; e < 16; ++e) bfr[e] = bp[e];
      acc2[nt] = __builtin_amdgcn_wmma_f32_16x16x32_bf16(
          false, a, false, bfr, (short)0, acc2[nt], false, false);
    }
  }

  // ---- layer 3: bias + leaky, dot with W3 (keep in f32), 16-lane reduce ----
  float p[8];
#pragma unroll
  for (int r = 0; r < 8; ++r) p[r] = 0.0f;
#pragma unroll
  for (int nt = 0; nt < 8; ++nt) {
    float bias = sB2[nt * 16 + lm];
    float w3v  = sW3[nt * 16 + lm];
#pragma unroll
    for (int r = 0; r < 8; ++r)
      p[r] += leaky(acc2[nt][r] + bias) * w3v;
  }
#pragma unroll
  for (int r = 0; r < 8; ++r) {
    p[r] += __shfl_xor(p[r], 1, 32);
    p[r] += __shfl_xor(p[r], 2, 32);
    p[r] += __shfl_xor(p[r], 4, 32);
    p[r] += __shfl_xor(p[r], 8, 32);
  }
  if (lm == 0) {             // lanes 0 (rows 0..7) and 16 (rows 8..15)
    float b3v = b3[0];
#pragma unroll
    for (int r = 0; r < 8; ++r) {
      int row = rowsBase + waveRow + r + kg * 8;
      if (row < N) logit[row] = p[r] + b3v;
    }
  }
}

__global__ void seg_max_kernel(const float* __restrict__ logit,
                               const long long* __restrict__ batch,
                               uint32_t* __restrict__ segmax, int N) {
  int n = blockIdx.x * blockDim.x + threadIdx.x;
  if (n < N) atomicMax(&segmax[(int)batch[n]], enc_f32(logit[n]));
}

__global__ void seg_exp_kernel(const float* __restrict__ logit,
                               const long long* __restrict__ batch,
                               const uint32_t* __restrict__ segmax,
                               float* __restrict__ evals,
                               float* __restrict__ segsum, int N) {
  int n = blockIdx.x * blockDim.x + threadIdx.x;
  if (n < N) {
    int g = (int)batch[n];
    float ev = expf(logit[n] - dec_f32(segmax[g]));
    evals[n] = ev;
    atomicAdd(&segsum[g], ev);
  }
}

__global__ void scores_kernel(const long long* __restrict__ batch,
                              const float* __restrict__ evals,
                              const float* __restrict__ segsum,
                              float* __restrict__ scores, int N) {
  int n = blockIdx.x * blockDim.x + threadIdx.x;
  if (n < N) scores[n] = evals[n] / segsum[(int)batch[n]];
}

// One block per graph; batch is sorted -> binary-search node range, atomic-free.
__global__ __launch_bounds__(256)
void pool_kernel(const float* __restrict__ h,
                 const long long* __restrict__ batch,
                 const float* __restrict__ evals,
                 const float* __restrict__ segsum,
                 float* __restrict__ hpool, int N) {
  __shared__ int sLo, sHi;
  const int g = blockIdx.x;
  if (threadIdx.x == 0) {
    long long key = g;
    int lo = 0, hi = N;
    while (lo < hi) { int mid = (lo + hi) >> 1; if (batch[mid] < key) lo = mid + 1; else hi = mid; }
    sLo = lo;
    int lo2 = lo, hi2 = N; long long key2 = key + 1;
    while (lo2 < hi2) { int mid = (lo2 + hi2) >> 1; if (batch[mid] < key2) lo2 = mid + 1; else hi2 = mid; }
    sHi = lo2;
  }
  __syncthreads();
  const int lo = sLo, hi = sHi, d = threadIdx.x;
  const float invsum = (hi > lo) ? 1.0f / segsum[g] : 0.0f;
  float acc = 0.0f;
  for (int n = lo; n < hi; ++n)
    acc += h[(size_t)n * D_IN + d] * (evals[n] * invsum);
  hpool[(size_t)g * D_IN + d] = acc;
}

extern "C" void kernel_launch(void* const* d_in, const int* in_sizes, int n_in,
                              void* d_out, int out_size, void* d_ws, size_t ws_size,
                              hipStream_t stream) {
  const float*     h     = (const float*)d_in[0];
  const long long* batch = (const long long*)d_in[1];
  const float*     W1    = (const float*)d_in[2];
  const float*     b1    = (const float*)d_in[3];
  const float*     W2    = (const float*)d_in[4];
  const float*     b2    = (const float*)d_in[5];
  const float*     W3    = (const float*)d_in[6];
  const float*     b3    = (const float*)d_in[7];
  const int N = in_sizes[0] / D_IN;

  float* hpool  = (float*)d_out;                          // [G, 256]
  float* scores = hpool + (size_t)NUM_GRAPHS * D_IN;      // [N]

  char* ws = (char*)d_ws;
  float*    logit  = (float*)ws;                          // N f32
  float*    evals  = logit + N;                           // N f32
  uint32_t* segmax = (uint32_t*)(evals + N);              // G u32
  float*    segsum = (float*)(segmax + NUM_GRAPHS);       // G f32
  uint16_t* w1f    = (uint16_t*)(segsum + NUM_GRAPHS);    // 32768 bf16
  uint16_t* w2f    = w1f + 32768;                         // 16384 bf16 (contiguous)

  prep_weights_kernel<<<(32768 + 255) / 256, 256, 0, stream>>>(W1, W2, w1f, w2f);
  seg_init_kernel<<<(NUM_GRAPHS + 255) / 256, 256, 0, stream>>>(segmax, segsum);

  const size_t smem =
      (size_t)(32768 + 16384 + 128 * H_PITCH + 128 * X_PITCH) * sizeof(uint16_t) +
      3 * 128 * sizeof(float);   // ~198 KB, fits 320 KB/WGP
  mlp_logit_kernel<<<(N + 127) / 128, 256, smem, stream>>>(
      h, b1, b2, W3, b3, w1f, logit, N);

  const int nb = (N + 255) / 256;
  seg_max_kernel<<<nb, 256, 0, stream>>>(logit, batch, segmax, N);
  seg_exp_kernel<<<nb, 256, 0, stream>>>(logit, batch, segmax, evals, segsum, N);
  scores_kernel<<<nb, 256, 0, stream>>>(batch, evals, segsum, scores, N);
  pool_kernel<<<NUM_GRAPHS, 256, 0, stream>>>(h, batch, evals, segsum, hpool, N);
}
